// LSTM_4389456576644
// MI455X (gfx1250) — matrix operand order repaired and checked by reading, not compiled, not run on hardware
//
#include <hip/hip_runtime.h>
#include <hip/hip_bf16.h>
#include <math.h>

// ---------------------------------------------------------------------------
// Problem constants: B=64 batch, T=512 seq, D=256 embed, H=256 hidden,
// 4H=1024 gate width, 20 classes.
// ---------------------------------------------------------------------------
#define BATCH 64
#define SEQ   512
#define EDIM  256
#define HID   256
#define FOURH 1024
#define NCLS  20
#define BT    (BATCH * SEQ)   // 32768 token rows
#define NWG   8               // persistent workgroups in the scan

typedef __attribute__((ext_vector_type(16))) __bf16         bf16x16;
typedef __attribute__((ext_vector_type(8)))  float          v8f;
typedef __attribute__((ext_vector_type(8)))  unsigned short u16x8;

union AFrag { bf16x16 v; u16x8 h[2]; };

__device__ __forceinline__ unsigned short f32_to_bf16_rne(float f) {
    unsigned int u = __float_as_uint(f);
    unsigned int r = u + 0x7FFFu + ((u >> 16) & 1u);
    return (unsigned short)(r >> 16);
}

// Device-scope barrier among NWG co-resident workgroups.
// Monotonic counter (reset by hipMemsetAsync each launch); target = NWG*phase.
__device__ __forceinline__ void global_barrier(unsigned int* ctr, unsigned int target) {
    __threadfence();          // make this thread's global stores visible (device scope)
    __syncthreads();          // all threads of the WG fenced before the arrive
    if (threadIdx.x == 0) {
        __hip_atomic_fetch_add(ctr, 1u, __ATOMIC_ACQ_REL, __HIP_MEMORY_SCOPE_AGENT);
        while (__hip_atomic_load(ctr, __ATOMIC_ACQUIRE, __HIP_MEMORY_SCOPE_AGENT) < target) {
            __builtin_amdgcn_s_sleep(2);
        }
    }
    __syncthreads();
}

// ---------------------------------------------------------------------------
// Kernel 1: transpose W,U [256,1024] -> bf16 N-major [1024,256] so WMMA
// B-fragments are two 16-byte per-lane loads.
// ---------------------------------------------------------------------------
__global__ void prep_weights(const float* __restrict__ W, const float* __restrict__ U,
                             unsigned short* __restrict__ Wt, unsigned short* __restrict__ Ut) {
    int idx = blockIdx.x * blockDim.x + threadIdx.x;
    if (idx < EDIM * FOURH) {
        int k = idx / FOURH;
        int n = idx % FOURH;
        Wt[n * EDIM + k] = f32_to_bf16_rne(W[idx]);
        Ut[n * HID  + k] = f32_to_bf16_rne(U[idx]);
    }
}

// ---------------------------------------------------------------------------
// Kernel 2: embedding gather + f32 -> bf16.  e_bf[BT, 256].
// (emb table is 51 MB -> L2-resident on the 192 MB L2.)
// ---------------------------------------------------------------------------
__global__ void embed_bf16(const int* __restrict__ x, const float* __restrict__ emb,
                           unsigned short* __restrict__ e) {
    int idx = blockIdx.x * blockDim.x + threadIdx.x;
    if (idx < BT * EDIM) {
        int r = idx >> 8;
        int d = idx & 255;
        int tok = x[r];
        e[idx] = f32_to_bf16_rne(emb[tok * EDIM + d]);
    }
}

// ---------------------------------------------------------------------------
// Kernel 3: big input projection  xz = e @ W + b, stored reordered [T, B, 4H].
// Block = 256 thr (8 waves) owns one 16-row M tile; wave w owns N tiles
// [8w, 8w+8).  K = 256 = 8 wmma steps.
// ---------------------------------------------------------------------------
__global__ __launch_bounds__(256) void input_proj(
    const unsigned short* __restrict__ A,    // e_bf [BT, 256]
    const unsigned short* __restrict__ Wt,   // [1024, 256] bf16 N-major
    const float* __restrict__ bias,          // [1024]
    float* __restrict__ xz)                  // [T, B, 1024]
{
    const int mtile = blockIdx.x;            // 0..2047
    const int w   = threadIdx.x >> 5;
    const int l   = threadIdx.x & 31;
    const int l15 = l & 15;
    const int lhi = l >> 4;

    const v8f vzero = {};
    v8f acc[8];
#pragma unroll
    for (int i = 0; i < 8; ++i) acc[i] = vzero;

    const int arow = mtile * 16 + l15;

#pragma unroll
    for (int ks = 0; ks < 8; ++ks) {
        AFrag a;
        const unsigned short* ap = A + arow * EDIM + ks * 32 + lhi * 8;
        a.h[0] = *(const u16x8*)(ap);
        a.h[1] = *(const u16x8*)(ap + 16);
#pragma unroll
        for (int nt = 0; nt < 8; ++nt) {
            const int col = (w * 8 + nt) * 16 + l15;
            const unsigned short* bp = Wt + col * EDIM + ks * 32 + lhi * 8;
            AFrag b;
            b.h[0] = *(const u16x8*)(bp);
            b.h[1] = *(const u16x8*)(bp + 16);
            acc[nt] = __builtin_amdgcn_wmma_f32_16x16x32_bf16(
                false, a.v, false, b.v, (short)0, acc[nt], false, false);
        }
    }

#pragma unroll
    for (int nt = 0; nt < 8; ++nt) {
        const int col = (w * 8 + nt) * 16 + l15;
        const float bv = bias[col];
#pragma unroll
        for (int g = 0; g < 8; ++g) {
            int r    = mtile * 16 + g + 8 * lhi;   // token row = b*512 + t
            int bidx = r >> 9;
            int tidx = r & 511;
            xz[(tidx * BATCH + bidx) * FOURH + col] = acc[nt][g] + bv;
        }
    }
}

// ---------------------------------------------------------------------------
// Kernel 4: multi-WGP LSTM scan.  Grid = 8 persistent WGs x 512 thr (16 waves).
// WG g owns hidden units [g*32, g*32+32) -> 128 gate columns (4 strips of 32).
// Its U^T slice (128x256 bf16 = 64 KB) is staged in LDS ONCE; per step all
// B-fragments are ds_load_b128.  h (64x256 bf16) is double-buffered in global
// (L2-hot); A-fragments are 2 global b128 loads per k-step.  z slice
// (64x128 f32 = 32 KB) lives in LDS.  One device barrier per step.
// ---------------------------------------------------------------------------
__global__ __launch_bounds__(512) void lstm_scan(
    const float* __restrict__ xz,            // [T, B, 1024]
    const unsigned short* __restrict__ Ut,   // [1024, 256] bf16 N-major
    const float* __restrict__ Wd,            // [256, 20]
    const float* __restrict__ bd,            // [20]
    unsigned short* __restrict__ hA,         // [64, 256] bf16 ping
    unsigned short* __restrict__ hB,         // [64, 256] bf16 pong
    float* __restrict__ hf,                  // [64, 256] f32 final h
    unsigned int* __restrict__ ctr,          // barrier counter (memset to 0)
    float* __restrict__ out)                 // [64, 20] softmax
{
    extern __shared__ char smem[];
    unsigned short* ut_s = (unsigned short*)smem;            // [128][256] bf16 = 64 KB
    float*          zs   = (float*)(smem + 128 * HID * 2);   // [64][128]  f32  = 32 KB

    const int g   = blockIdx.x;              // 0..7, hidden slice owner
    const int tid = threadIdx.x;
    const int w   = tid >> 5;
    const int l   = tid & 31;
    const int l15 = l & 15;
    const int lhi = l >> 4;
    const int m     = w & 3;                 // M tile: batch rows m*16..+15
    const int npair = w >> 2;                // 0..3 -> local N tiles 2*npair, +1

    // gate-phase ownership
    const int ul = tid & 31;                 // hidden unit within slice
    // bb = j*16 + w for j=0..3

    // ---- stage U^T slice into LDS (once) ----
    for (int i = tid; i < 128 * (HID / 8); i += 512) {       // 4096 x 16B chunks
        int lc = i >> 5;                     // local col 0..127
        int ck = i & 31;                     // 16B chunk within row
        int q  = lc >> 5;                    // gate 0..3
        int within = lc & 31;
        int utrow  = q * 256 + g * 32 + within;
        *(u16x8*)(ut_s + lc * HID + ck * 8) = *(const u16x8*)(Ut + utrow * HID + ck * 8);
    }

    // ---- zero this WG's slice of h ping buffer ----
    for (int j = 0; j < 4; ++j) {
        int bb = j * 16 + w;
        hA[bb * HID + g * 32 + ul] = 0;
    }
    float c[4], hsave[4];
#pragma unroll
    for (int j = 0; j < 4; ++j) { c[j] = 0.f; hsave[j] = 0.f; }

    unsigned int phase = 1;
    global_barrier(ctr, NWG * phase);        // h zeroed + LDS staged everywhere

    const unsigned short* hc = hA;
    unsigned short*       hn = hB;

#pragma unroll 1
    for (int t = 0; t < SEQ; ++t) {
        const float* xzt = xz + (size_t)t * BATCH * FOURH;

        // ---------- GEMM phase: z_slice = xz_t[:, cols] + h @ U[:, cols] ----------
        v8f acc[2];
#pragma unroll
        for (int i = 0; i < 2; ++i) {
            const int nt  = npair * 2 + i;                     // local N tile 0..7
            const int q   = nt >> 1;
            const int col = q * 256 + g * 32 + (nt & 1) * 16 + l15;
#pragma unroll
            for (int gg = 0; gg < 8; ++gg) {
                int row = m * 16 + gg + 8 * lhi;
                acc[i][gg] = xzt[row * FOURH + col];
            }
        }
#pragma unroll
        for (int ks = 0; ks < 8; ++ks) {
            AFrag a;
            const unsigned short* ap = hc + (m * 16 + l15) * HID + ks * 32 + lhi * 8;
            a.h[0] = *(const u16x8*)(ap);
            a.h[1] = *(const u16x8*)(ap + 16);
#pragma unroll
            for (int i = 0; i < 2; ++i) {
                const int nt = npair * 2 + i;
                const unsigned short* bp = ut_s + (nt * 16 + l15) * HID + ks * 32 + lhi * 8;
                AFrag b;
                b.h[0] = *(const u16x8*)(bp);
                b.h[1] = *(const u16x8*)(bp + 16);
                acc[i] = __builtin_amdgcn_wmma_f32_16x16x32_bf16(
                    false, a.v, false, b.v, (short)0, acc[i], false, false);
            }
        }
#pragma unroll
        for (int i = 0; i < 2; ++i) {
            const int nt = npair * 2 + i;
#pragma unroll
            for (int gg = 0; gg < 8; ++gg) {
                int row = m * 16 + gg + 8 * lhi;
                zs[row * 128 + nt * 16 + l15] = acc[i][gg];
            }
        }

        // prefetch next timestep's 256 KB xz slab (8 WGs x 512 thr x 64 B)
        if (t + 1 < SEQ) {
            __builtin_prefetch(xz + (size_t)(t + 1) * BATCH * FOURH + (g * 512 + tid) * 16, 0, 1);
        }

        __syncthreads();                     // z slice complete in LDS

        // ---------- gate phase ----------
#pragma unroll
        for (int j = 0; j < 4; ++j) {
            const int bb = j * 16 + w;
            float zi = zs[bb * 128 + ul];
            float zf = zs[bb * 128 + 32 + ul];
            float zg = zs[bb * 128 + 64 + ul];
            float zo = zs[bb * 128 + 96 + ul];
            float ig = 1.f / (1.f + __expf(-zi));
            float fg = 1.f / (1.f + __expf(-zf));
            float gg = tanhf(zg);
            float og = 1.f / (1.f + __expf(-zo));
            float cn = fg * c[j] + ig * gg;
            c[j] = cn;
            float hn_v = og * tanhf(cn);
            hsave[j] = hn_v;
            hn[bb * HID + g * 32 + ul] = f32_to_bf16_rne(hn_v);
        }
        if (t == SEQ - 1) {
#pragma unroll
            for (int j = 0; j < 4; ++j) {
                const int bb = j * 16 + w;
                hf[bb * HID + g * 32 + ul] = hsave[j];
            }
        }

        phase++;
        global_barrier(ctr, NWG * phase);    // h(t+1) fully written device-wide

        // swap ping/pong
        const unsigned short* tmp = hc; hc = hn; hn = (unsigned short*)tmp;
    }

    // ---------- final dense + softmax: WG 0 only ----------
    if (g != 0) return;

    float* logits = zs;                      // reuse LDS (1280 floats)
    for (int i = tid; i < BATCH * NCLS; i += 512) {
        int bb  = i / NCLS;
        int cls = i % NCLS;
        float s = bd[cls];
        for (int k = 0; k < HID; ++k) s += hf[bb * HID + k] * Wd[k * NCLS + cls];
        logits[i] = s;
    }
    __syncthreads();

    if (tid < BATCH) {
        float mx = -1e30f;
        for (int i = 0; i < NCLS; ++i) mx = fmaxf(mx, logits[tid * NCLS + i]);
        float ex[NCLS];
        float sum = 0.f;
        for (int i = 0; i < NCLS; ++i) {
            ex[i] = __expf(logits[tid * NCLS + i] - mx);
            sum += ex[i];
        }
        float inv = 1.f / sum;
        for (int i = 0; i < NCLS; ++i) out[tid * NCLS + i] = ex[i] * inv;
    }
}

// ---------------------------------------------------------------------------
// Host-side launcher
// ---------------------------------------------------------------------------
extern "C" void kernel_launch(void* const* d_in, const int* in_sizes, int n_in,
                              void* d_out, int out_size, void* d_ws, size_t ws_size,
                              hipStream_t stream) {
    (void)in_sizes; (void)n_in; (void)out_size; (void)ws_size;

    const int*   x   = (const int*)d_in[0];
    const float* emb = (const float*)d_in[1];
    const float* W   = (const float*)d_in[2];
    const float* U   = (const float*)d_in[3];
    const float* b   = (const float*)d_in[4];
    const float* Wd  = (const float*)d_in[5];
    const float* bd  = (const float*)d_in[6];
    float* out = (float*)d_out;

    // workspace layout
    char* ws = (char*)d_ws;
    size_t off = 0;
    unsigned short* e_bf = (unsigned short*)(ws + off);  off += (size_t)BT * EDIM * 2;    // 16 MiB
    unsigned short* Wt   = (unsigned short*)(ws + off);  off += (size_t)EDIM * FOURH * 2; // 512 KiB
    unsigned short* Ut   = (unsigned short*)(ws + off);  off += (size_t)HID  * FOURH * 2; // 512 KiB
    float* xz            = (float*)(ws + off);           off += (size_t)SEQ * BATCH * FOURH * 4; // 128 MiB
    unsigned short* hA   = (unsigned short*)(ws + off);  off += (size_t)BATCH * HID * 2;
    unsigned short* hB   = (unsigned short*)(ws + off);  off += (size_t)BATCH * HID * 2;
    float* hf            = (float*)(ws + off);           off += (size_t)BATCH * HID * 4;
    unsigned int* ctr    = (unsigned int*)(ws + off);    off += 64;

    // reset barrier counter (graph-capturable, deterministic per replay)
    hipMemsetAsync(ctr, 0, 64, stream);

    prep_weights<<<(EDIM * FOURH + 255) / 256, 256, 0, stream>>>(W, U, Wt, Ut);
    embed_bf16 <<<(BT * EDIM + 255) / 256,   256, 0, stream>>>(x, emb, e_bf);
    input_proj <<<BT / 16,                   256, 0, stream>>>(e_bf, Wt, b, xz);

    size_t smem = (size_t)128 * HID * 2 + (size_t)BATCH * 128 * 4;   // 64 KiB + 32 KiB
    lstm_scan  <<<NWG, 512, smem, stream>>>(xz, Ut, Wd, bd, hA, hB, hf, ctr, out);
}
